// SemiDualOptimalTransportLayer_43147241455739
// MI455X (gfx1250) — matrix operand order repaired and checked by reading, not compiled, not run on hardware
//
#include <hip/hip_runtime.h>
#include <stdint.h>

typedef __attribute__((ext_vector_type(16))) __bf16 v16bf;
typedef __attribute__((ext_vector_type(8)))  float  v8f;

#define TILE_M 128
#define TILE_N 128
#define TILE_K 32
#define MAX_D  256
#define XSTRIDE (MAX_D + 8)   // halves: 528B rows -> 16B aligned, 4-bank shift/row, conflict-free b128
#define YSTRIDE 40            // halves: 80B rows  -> 16B aligned, 20-bank shift/row, conflict-free b128

__device__ __forceinline__ unsigned short f32_bf16_rne(float f) {
  unsigned int u = __float_as_uint(f);
  unsigned int r = 0x7FFFu + ((u >> 16) & 1u);
  return (unsigned short)((u + r) >> 16);
}
__device__ __forceinline__ float bf16_f32(unsigned short h) {
  return __uint_as_float(((unsigned int)h) << 16);
}

// ---------------- prep: Y fp32 -> bf16 hi/lo ----------------
__global__ void prep_convert_y(const float* __restrict__ Y,
                               unsigned short* __restrict__ yH,
                               unsigned short* __restrict__ yL, int total4) {
  int idx = blockIdx.x * blockDim.x + threadIdx.x;
  if (idx >= total4) return;
  const float4 v = ((const float4*)Y)[idx];
  float e[4] = {v.x, v.y, v.z, v.w};
  unsigned short h[4], l[4];
#pragma unroll
  for (int k = 0; k < 4; ++k) {
    h[k] = f32_bf16_rne(e[k]);
    l[k] = f32_bf16_rne(e[k] - bf16_f32(h[k]));
  }
  uint2 ph = make_uint2((unsigned)h[0] | ((unsigned)h[1] << 16),
                        (unsigned)h[2] | ((unsigned)h[3] << 16));
  uint2 pl = make_uint2((unsigned)l[0] | ((unsigned)l[1] << 16),
                        (unsigned)l[2] | ((unsigned)l[3] << 16));
  ((uint2*)yH)[idx] = ph;
  ((uint2*)yL)[idx] = pl;
}

// ---------------- prep: row stats (0.5*||row||^2 [- psi]) ----------------
__global__ void prep_rowstats(const float* __restrict__ A, const float* __restrict__ psi,
                              float* __restrict__ out, int rows, int D, int hasPsi) {
  int r = blockIdx.x * blockDim.x + threadIdx.x;
  if (r >= rows) return;
  const float4* p = (const float4*)(A + (size_t)r * D);
  float s = 0.f;
  for (int k = 0; k < D / 4; ++k) {
    float4 v = p[k];
    s += v.x * v.x + v.y * v.y + v.z * v.z + v.w * v.w;
  }
  float t = 0.5f * s;
  if (hasPsi) t -= psi[r];
  out[r] = t;
}

// ---------------- fused GEMM + row-min ----------------
// 256 threads = 8 waves. Block owns TILE_M rows of X (converted once into resident LDS),
// sweeps all M columns in TILE_N slabs, K streamed in TILE_K chunks.
// Y tiles arrive via GLOBAL_LOAD_ASYNC_TO_LDS_B128 (ASYNCcnt), double-buffered.
// Wave w owns rows [w*16, w*16+16) exclusively -> no atomics on output.
__global__ void __launch_bounds__(256)
otp_rowmin_kernel(const float* __restrict__ X,
                  const unsigned short* __restrict__ yH,
                  const unsigned short* __restrict__ yL,
                  const float* __restrict__ tcol,
                  const float* __restrict__ sx,
                  float* __restrict__ rowmin,
                  int N, int M, int D) {
  __shared__ __align__(16) unsigned short XsH[TILE_M * XSTRIDE];              // 66 KB
  __shared__ __align__(16) unsigned short XsL[TILE_M * XSTRIDE];              // 66 KB
  __shared__ __align__(16) unsigned short Ys[2][2][TILE_N * YSTRIDE];         // 40 KB

  const int tid  = threadIdx.x;
  const int lane = tid & 31;
  const int wave = tid >> 5;      // 0..7
  const int l15  = lane & 15;
  const int lhi  = lane >> 4;     // 0/1
  const int rowBlk = blockIdx.x * TILE_M;

  // ---- prologue: convert this block's X tile to bf16 hi/lo, once ----
#pragma unroll 4
  for (int it = 0; it < (TILE_M * MAX_D / 4) / 256; ++it) {   // 32 iters, one float4 each
    int idx = it * 256 + tid;
    int row = idx >> 6;          // D/4 = 64 float4 per row
    int c4  = idx & 63;
    const float4 v = *(const float4*)(X + (size_t)(rowBlk + row) * D + c4 * 4);
    float e[4] = {v.x, v.y, v.z, v.w};
    unsigned short h[4], l[4];
#pragma unroll
    for (int k = 0; k < 4; ++k) {
      h[k] = f32_bf16_rne(e[k]);
      l[k] = f32_bf16_rne(e[k] - bf16_f32(h[k]));
    }
    uint2 ph = make_uint2((unsigned)h[0] | ((unsigned)h[1] << 16),
                          (unsigned)h[2] | ((unsigned)h[3] << 16));
    uint2 pl = make_uint2((unsigned)l[0] | ((unsigned)l[1] << 16),
                          (unsigned)l[2] | ((unsigned)l[3] << 16));
    *(uint2*)&XsH[row * XSTRIDE + c4 * 4] = ph;
    *(uint2*)&XsL[row * XSTRIDE + c4 * 4] = pl;
  }

  const int nKT = D / TILE_K;
  const int nSteps = (M / TILE_N) * nKT;

  // async DMA of one (jt,kt) Y panel (bf16 hi+lo) into LDS buffer (step&1)
  auto issueY = [&](int s) {
    int jt = s / nKT;
    int kt = s - jt * nKT;
    int buf = s & 1;
    const unsigned short* baseH = yH + (size_t)(jt * TILE_N) * D + kt * TILE_K;
    const unsigned short* baseL = yL + (size_t)(jt * TILE_N) * D + kt * TILE_K;
#pragma unroll
    for (int it = 0; it < 2; ++it) {
      int idx = it * 256 + tid;
      int row = idx >> 2;        // 4 x 16B chunks per row
      int c8  = idx & 3;
      unsigned voff = (unsigned)((row * D + c8 * 8) * 2);   // bytes
      unsigned dH = (unsigned)(size_t)&Ys[buf][0][row * YSTRIDE + c8 * 8];
      unsigned dL = (unsigned)(size_t)&Ys[buf][1][row * YSTRIDE + c8 * 8];
      asm volatile("global_load_async_to_lds_b128 %0, %1, %2"
                   :: "v"(dH), "v"(voff), "s"(baseH) : "memory");
      asm volatile("global_load_async_to_lds_b128 %0, %1, %2"
                   :: "v"(dL), "v"(voff), "s"(baseL) : "memory");
    }
  };

  float rmin[8];
#pragma unroll
  for (int r = 0; r < 8; ++r) rmin[r] = 3.0e38f;
  const v8f vzero = {0.f, 0.f, 0.f, 0.f, 0.f, 0.f, 0.f, 0.f};
  v8f acc[8];

  issueY(0);
  for (int step = 0; step < nSteps; ++step) {
    const int buf = step & 1;
    const int jt = step / nKT;
    const int kt = step - jt * nKT;

    // my async writes done -> barrier makes everyone's writes (and last step's reads) visible
    asm volatile("s_wait_asynccnt 0x0" ::: "memory");
    __syncthreads();
    if (step + 1 < nSteps) issueY(step + 1);   // prefetch next panel into other buffer

    if (kt == 0) {
#pragma unroll
      for (int nt = 0; nt < 8; ++nt) acc[nt] = vzero;
    }

    // A fragment (ISA 16-bit A layout: lane group g holds K = {g*8..+7} U {16+g*8..+7})
    union Frag { v16bf bf; uint4 q[2]; } aH, aL, bH, bL;
    const int abase = (wave * 16 + l15) * XSTRIDE + kt * TILE_K + lhi * 8;
    aH.q[0] = *(const uint4*)&XsH[abase];
    aH.q[1] = *(const uint4*)&XsH[abase + 16];
    aL.q[0] = *(const uint4*)&XsL[abase];
    aL.q[1] = *(const uint4*)&XsL[abase + 16];

    const unsigned short* ysH = Ys[buf][0];
    const unsigned short* ysL = Ys[buf][1];
#pragma unroll
    for (int nt = 0; nt < 8; ++nt) {
      // B fragment (lane group g holds K = {g*8..+7} U {16+g*8..+7} of column l15, matching A)
      const int bbase = (nt * 16 + l15) * YSTRIDE + lhi * 8;
      bH.q[0] = *(const uint4*)&ysH[bbase];
      bH.q[1] = *(const uint4*)&ysH[bbase + 16];
      bL.q[0] = *(const uint4*)&ysL[bbase];
      bL.q[1] = *(const uint4*)&ysL[bbase + 16];
      // split-precision fp32-accurate product: xh*yh + xh*yl + xl*yh
      acc[nt] = __builtin_amdgcn_wmma_f32_16x16x32_bf16(false, aH.bf, false, bH.bf,
                                                        (short)0, acc[nt], false, false);
      acc[nt] = __builtin_amdgcn_wmma_f32_16x16x32_bf16(false, aH.bf, false, bL.bf,
                                                        (short)0, acc[nt], false, false);
      acc[nt] = __builtin_amdgcn_wmma_f32_16x16x32_bf16(false, aL.bf, false, bH.bf,
                                                        (short)0, acc[nt], false, false);
    }

    if (kt == nKT - 1) {
      // fold slab into running row-min:
      // min_j(cost - psi) = 0.5||x||^2 + min_j( (0.5||y||^2 - psi_j) - x.y )
#pragma unroll
      for (int nt = 0; nt < 8; ++nt) {
        float tj = tcol[jt * TILE_N + nt * 16 + l15];
#pragma unroll
        for (int r = 0; r < 8; ++r)
          rmin[r] = fminf(rmin[r], tj - acc[nt][r]);
      }
    }
  }

  // C layout: VGPR r, lane l15 -> (row = 16*wave + 8*lhi + r, col = l15).
  // Min across the 16 lanes sharing a row, then add 0.5||x||^2 and store (exclusive rows).
#pragma unroll
  for (int r = 0; r < 8; ++r) {
    float v = rmin[r];
#pragma unroll
    for (int off = 8; off >= 1; off >>= 1)
      v = fminf(v, __shfl_xor(v, off, 16));
    if (l15 == 0) {
      int rowg = rowBlk + wave * 16 + lhi * 8 + r;
      rowmin[rowg] = v + sx[rowg];
    }
  }
}

// ---------------- final scalar reduction ----------------
__global__ void final_reduce(const float* __restrict__ rowmin, const float* __restrict__ psi,
                             float* __restrict__ out, int N, int M) {
  __shared__ float red[256];
  float s = 0.f;
  for (int i = threadIdx.x; i < N; i += 256) s += rowmin[i];
  float sp = 0.f;
  for (int j = threadIdx.x; j < M; j += 256) sp += psi[j];
  red[threadIdx.x] = s / (float)N + sp / (float)M;
  __syncthreads();
  for (int st = 128; st > 0; st >>= 1) {
    if (threadIdx.x < st) red[threadIdx.x] += red[threadIdx.x + st];
    __syncthreads();
  }
  if (threadIdx.x == 0) out[0] = red[0];
}

extern "C" void kernel_launch(void* const* d_in, const int* in_sizes, int n_in,
                              void* d_out, int out_size, void* d_ws, size_t ws_size,
                              hipStream_t stream) {
  const float* X   = (const float*)d_in[0];
  const float* Y   = (const float*)d_in[1];
  const float* psi = (const float*)d_in[2];
  const int M = in_sizes[2];
  const int D = in_sizes[1] / M;
  const int N = in_sizes[0] / D;

  char* ws = (char*)d_ws;
  unsigned short* yH = (unsigned short*)ws;  ws += (size_t)M * D * sizeof(unsigned short);
  unsigned short* yL = (unsigned short*)ws;  ws += (size_t)M * D * sizeof(unsigned short);
  float* tcol   = (float*)ws;                ws += (size_t)M * sizeof(float);
  float* sx     = (float*)ws;                ws += (size_t)N * sizeof(float);
  float* rowmin = (float*)ws;

  prep_convert_y<<<(M * D / 4 + 255) / 256, 256, 0, stream>>>(Y, yH, yL, M * D / 4);
  prep_rowstats<<<(M + 255) / 256, 256, 0, stream>>>(Y, psi, tcol, M, D, 1);
  prep_rowstats<<<(N + 255) / 256, 256, 0, stream>>>(X, psi, sx, N, D, 0);
  otp_rowmin_kernel<<<N / TILE_M, 256, 0, stream>>>(X, yH, yL, tcol, sx, rowmin, N, M, D);
  final_reduce<<<1, 256, 0, stream>>>(rowmin, psi, (float*)d_out, N, M);
}
